// ODEFunc_51994874085478
// MI455X (gfx1250) — matrix operand order repaired and checked by reading, not compiled, not run on hardware
//
#include <hip/hip_runtime.h>

typedef __attribute__((ext_vector_type(16))) _Float16 v16h;
typedef __attribute__((ext_vector_type(8)))  float    v8f;

#define BB   1024
#define TT   200
#define XSD  32
#define ASD  32
#define XHD  64
#define AHD  64
#define ROWS 16        // batch rows per workgroup == WMMA M tile
#define NTHREADS 128   // 4 waves (wave32)

// LDS arena size in bytes — must match the carve order inside the kernel.
//  f16 region: 72704 halves = 145408 B ; f32 region: 21104 floats = 84416 B
#define SMEM_BYTES 229824

__device__ __forceinline__ float sigm(float v) { return 1.f / (1.f + __expf(-v)); }

// ---- A fragment: 16x32 f16, activations staged row-major [M][lda] in LDS ----
// ISA 16-bit A layout: lanes 0-15 M=l, K base 0; lanes 16-31 M=l-16, K base 8.
// VGPR v<4 -> K = kb+2v ; v>=4 -> K = kb+2v+8  (consecutive pairs -> b32 LDS reads)
__device__ __forceinline__ v16h load_A(const _Float16* a, int lda, int k0) {
  const int lane = threadIdx.x & 31;
  const int m  = lane & 15;
  const int kb = (lane < 16) ? 0 : 8;
  const _Float16* p = a + (size_t)m * lda + k0;
  v16h r;
#pragma unroll
  for (int v = 0; v < 8; ++v) {
    const int k = (v < 4) ? (kb + 2 * v) : (kb + 2 * v + 8);
    r[2 * v]     = p[k];
    r[2 * v + 1] = p[k + 1];
  }
  return r;
}

// ---- B fragment: 32x16 f16, weights stored [N][K] row-major (torch layout) ----
// lanes 0-15: N=l, K 0..7,16..23 ; lanes 16-31: N=l-16, K 8..15,24..31 (matches A).
__device__ __forceinline__ v16h load_B(const _Float16* w, int ldk, int n0, int k0) {
  const int lane = threadIdx.x & 31;
  const int n  = lane & 15;
  const int kb = (lane < 16) ? 0 : 8;
  const _Float16* p = w + (size_t)(n0 + n) * ldk + k0 + kb;
  v16h r;
#pragma unroll
  for (int v = 0; v < 16; ++v) r[v] = p[(v < 8) ? v : v + 8];
  return r;
}

// ---- D accumulator 16x16 f32 -> LDS row-major [M][ldn] ----
// VGPR r: lanes 0-15 -> M=r ; lanes 16-31 -> M=r+8 ; column = n0 + (lane&15)
__device__ __forceinline__ void store_D(float* d, int ldn, int n0, v8f acc) {
  const int lane = threadIdx.x & 31;
  const int n  = lane & 15;
  const int mo = (lane < 16) ? 0 : 8;
#pragma unroll
  for (int r = 0; r < 8; ++r) d[(size_t)(mo + r) * ldn + n0 + n] = acc[r];
}

// D(16x16) = A(16xK) * W[n0:n0+16, 0:K]^T + bias, K-loop of 16x16x32 WMMAs
__device__ __forceinline__ v8f mm_tile(const _Float16* A, int lda,
                                       const _Float16* W, int ldk,
                                       int n0, int K, const float* bias) {
  const int lane = threadIdx.x & 31;
  const float bv = bias[n0 + (lane & 15)];
  v8f c;
#pragma unroll
  for (int r = 0; r < 8; ++r) c[r] = bv;
  for (int k0 = 0; k0 < K; k0 += 32) {
    v16h a = load_A(A, lda, k0);
    v16h b = load_B(W, ldk, n0, k0);
    c = __builtin_amdgcn_wmma_f32_16x16x32_f16(false, a, false, b, (short)0, c,
                                               false, false);
  }
  return c;
}

__global__ void __launch_bounds__(NTHREADS, 1)
node_gru_ode_kernel(const float* __restrict__ x, const float* __restrict__ y,
                    const float* __restrict__ ts,
                    const float* a_wih, const float* a_bih,
                    const float* a_whh, const float* a_bhh,
                    const float* x_wih, const float* x_bih,
                    const float* x_whh, const float* x_bhh,
                    const float* zx_w1, const float* zx_b1,
                    const float* zx_w2, const float* zx_b2,
                    const float* za_w1, const float* za_b1,
                    const float* za_w2, const float* za_b2,
                    const float* dx_w1, const float* dx_b1,
                    const float* dx_w2, const float* dx_b2,
                    float* __restrict__ out) {
  extern __shared__ char smem[];
  char* sp = smem;
  // ---- f16 region (all counts even -> alignment preserved) ----
  _Float16* Wa_ih = (_Float16*)sp; sp += 192 * 64 * 2;
  _Float16* Wa_hh = (_Float16*)sp; sp += 192 * 64 * 2;
  _Float16* Wx_ih = (_Float16*)sp; sp += 192 * 32 * 2;
  _Float16* Wx_hh = (_Float16*)sp; sp += 192 * 64 * 2;
  _Float16* Wzx1  = (_Float16*)sp; sp += 64 * 64 * 2;
  _Float16* Wzx2  = (_Float16*)sp; sp += 64 * 64 * 2;
  _Float16* Wza1  = (_Float16*)sp; sp += 64 * 64 * 2;
  _Float16* Wza2  = (_Float16*)sp; sp += 64 * 64 * 2;
  _Float16* Wdx1  = (_Float16*)sp; sp += 64 * 128 * 2;
  _Float16* Wdx2  = (_Float16*)sp; sp += 32 * 64 * 2;
  _Float16* AFS   = (_Float16*)sp; sp += 16 * 192 * 2;   // f16 activation staging
  // ---- f32 region ----
  float* Bai  = (float*)sp; sp += 192 * 4;
  float* Bah  = (float*)sp; sp += 192 * 4;
  float* Bxi  = (float*)sp; sp += 192 * 4;
  float* Bxh  = (float*)sp; sp += 192 * 4;
  float* Bzx1 = (float*)sp; sp += 64 * 4;
  float* Bzx2 = (float*)sp; sp += 64 * 4;
  float* Bza1 = (float*)sp; sp += 64 * 4;
  float* Bza2 = (float*)sp; sp += 64 * 4;
  float* Bdx1 = (float*)sp; sp += 64 * 4;
  float* Bdx2 = (float*)sp; sp += 32 * 4;
  float* U  = (float*)sp; sp += 16 * 160 * 4;  // state: zx|za|x per row
  float* UT = (float*)sp; sp += 16 * 160 * 4;  // RK4 eval input
  float* KC = (float*)sp; sp += 16 * 160 * 4;  // current k
  float* KA = (float*)sp; sp += 16 * 160 * 4;  // k accumulator
  float* GI = (float*)sp; sp += 16 * 192 * 4;  // gate gi
  float* GH = (float*)sp; sp += 16 * 192 * 4;  // gate gh
  float* HH = (float*)sp; sp += 16 * 192 * 4;  // MLP layer-1 out hzx|hza|hdx
  float* AT = (float*)sp; sp += 16 * 32 * 4;   // a_t observation
  float* MK = (float*)sp; sp += 16 * 4;        // obs mask

  const int tid   = threadIdx.x;
  const int wave  = tid >> 5;
  const int bBase = blockIdx.x * ROWS;

  // ---- load all weights once: f32 HBM -> f16 LDS (stays resident for 199 steps) ----
  for (int i = tid; i < 192 * 64; i += NTHREADS) {
    Wa_ih[i] = (_Float16)a_wih[i];
    Wa_hh[i] = (_Float16)a_whh[i];
    Wx_hh[i] = (_Float16)x_whh[i];
  }
  for (int i = tid; i < 192 * 32; i += NTHREADS) Wx_ih[i] = (_Float16)x_wih[i];
  for (int i = tid; i < 64 * 64; i += NTHREADS) {
    Wzx1[i] = (_Float16)zx_w1[i]; Wzx2[i] = (_Float16)zx_w2[i];
    Wza1[i] = (_Float16)za_w1[i]; Wza2[i] = (_Float16)za_w2[i];
  }
  for (int i = tid; i < 64 * 128; i += NTHREADS) Wdx1[i] = (_Float16)dx_w1[i];
  for (int i = tid; i < 32 * 64; i += NTHREADS) Wdx2[i] = (_Float16)dx_w2[i];
  for (int i = tid; i < 192; i += NTHREADS) {
    Bai[i] = a_bih[i]; Bah[i] = a_bhh[i]; Bxi[i] = x_bih[i]; Bxh[i] = x_bhh[i];
  }
  for (int i = tid; i < 64; i += NTHREADS) {
    Bzx1[i] = zx_b1[i]; Bzx2[i] = zx_b2[i];
    Bza1[i] = za_b1[i]; Bza2[i] = za_b2[i]; Bdx1[i] = dx_b1[i];
  }
  for (int i = tid; i < 32; i += NTHREADS) Bdx2[i] = dx_b2[i];

  // ---- init state: zx=za=0, x = x[:,0,:]; emit out[:,0,:] ----
  for (int i = tid; i < 16 * 160; i += NTHREADS) U[i] = 0.f;
  __syncthreads();
  for (int i = tid; i < 16 * 32; i += NTHREADS) {
    int m = i >> 5, j = i & 31;
    float v = x[((size_t)(bBase + m) * TT) * XSD + j];
    U[m * 160 + 128 + j] = v;
    out[((size_t)(bBase + m) * TT) * XSD + j] = v;
  }
  __syncthreads();

  for (int t = 0; t < TT - 1; ++t) {
    const float dt = ts[t + 1] - ts[t];

    // ---- stage old state + observation as f16 A-operands ----
    // AFS[:,0:32]=x_c, [:,32:64]=a_t, [:,64:128]=za, [:,128:192]=zx
    for (int i = tid; i < 16 * 32; i += NTHREADS) {
      int m = i >> 5, j = i & 31;
      float a = y[((size_t)(bBase + m) * TT + t) * ASD + j];
      AT[m * 32 + j]        = a;
      AFS[m * 192 + 32 + j] = (_Float16)a;
      AFS[m * 192 + j]      = (_Float16)U[m * 160 + 128 + j];
    }
    for (int i = tid; i < 16 * 64; i += NTHREADS) {
      int m = i >> 6, j = i & 63;
      AFS[m * 192 + 64 + j]  = (_Float16)U[m * 160 + 64 + j];
      AFS[m * 192 + 128 + j] = (_Float16)U[m * 160 + j];
    }
    __syncthreads();
    if (tid < 16) {
      float s = 0.f;
      for (int j = 0; j < 32; ++j) s += AT[tid * 32 + j];
      MK[tid] = (s != 0.f) ? 1.f : 0.f;
    }
    // prefetch next step's observation rows into cache (global_prefetch_b8)
    {
      const float* pf = &y[((size_t)(bBase + (tid & 15)) * TT + (t + 1)) * ASD];
      __builtin_prefetch(pf, 0, 3);
    }

    // ---- a-GRU: gi = [x_c|a_t](16x64) @ Wa_ih^T, gh = za @ Wa_hh^T (192 wide) ----
    for (int j = 0; j < 3; ++j) {
      const int n0 = 16 * wave + 64 * j;
      store_D(GI, 192, n0, mm_tile(AFS,      192, Wa_ih, 64, n0, 64, Bai));
      store_D(GH, 192, n0, mm_tile(AFS + 64, 192, Wa_hh, 64, n0, 64, Bah));
    }
    __syncthreads();
    for (int i = tid; i < 16 * 64; i += NTHREADS) {
      int m = i >> 6, j = i & 63;
      float r  = sigm(GI[m * 192 + j] + GH[m * 192 + j]);
      float z  = sigm(GI[m * 192 + 64 + j] + GH[m * 192 + 64 + j]);
      float nn = tanhf(GI[m * 192 + 128 + j] + r * GH[m * 192 + 128 + j]);
      float h  = U[m * 160 + 64 + j];
      float hn = (1.f - z) * nn + z * h;
      U[m * 160 + 64 + j] = (MK[m] != 0.f) ? hn : h;
    }
    __syncthreads();

    // ---- x-GRU: gi = x_c(16x32) @ Wx_ih^T (K=32), gh = zx @ Wx_hh^T ----
    for (int j = 0; j < 3; ++j) {
      const int n0 = 16 * wave + 64 * j;
      store_D(GI, 192, n0, mm_tile(AFS,       192, Wx_ih, 32, n0, 32, Bxi));
      store_D(GH, 192, n0, mm_tile(AFS + 128, 192, Wx_hh, 64, n0, 64, Bxh));
    }
    __syncthreads();
    for (int i = tid; i < 16 * 64; i += NTHREADS) {
      int m = i >> 6, j = i & 63;
      float r  = sigm(GI[m * 192 + j] + GH[m * 192 + j]);
      float z  = sigm(GI[m * 192 + 64 + j] + GH[m * 192 + 64 + j]);
      float nn = tanhf(GI[m * 192 + 128 + j] + r * GH[m * 192 + 128 + j]);
      float h  = U[m * 160 + j];
      U[m * 160 + j] = (1.f - z) * nn + z * h;
    }
    __syncthreads();

    // ---- RK4 over u = [zx|za|x] (160 per row) ----
    for (int e = 0; e < 4; ++e) {
      const float* src = (e == 0) ? U : UT;
      // stage z = [zx|za] (cols 0..127) as f16
      for (int i = tid; i < 16 * 128; i += NTHREADS) {
        int m = i >> 7, c2 = i & 127;
        AFS[m * 192 + c2] = (_Float16)src[m * 160 + c2];
      }
      __syncthreads();
      // layer 1: hzx, hza (K=64) ; hdx (K=128, input = full z)
      {
        const int n0 = 16 * wave;
        store_D(HH, 192, n0,       mm_tile(AFS,      192, Wzx1,  64, n0,  64, Bzx1));
        store_D(HH, 192, 64 + n0,  mm_tile(AFS + 64, 192, Wza1,  64, n0,  64, Bza1));
        store_D(HH, 192, 128 + n0, mm_tile(AFS,      192, Wdx1, 128, n0, 128, Bdx1));
      }
      __syncthreads();
      // leaky_relu fused into f32 -> f16 restage
      for (int i = tid; i < 16 * 192; i += NTHREADS) {
        float v = HH[i];
        AFS[i] = (_Float16)(v > 0.f ? v : 0.01f * v);
      }
      __syncthreads();
      // layer 2: dzx, dza (64 wide) ; dx (32 wide -> waves 0,1)
      {
        const int n0 = 16 * wave;
        store_D(KC, 160, n0,      mm_tile(AFS,      192, Wzx2, 64, n0, 64, Bzx2));
        store_D(KC, 160, 64 + n0, mm_tile(AFS + 64, 192, Wza2, 64, n0, 64, Bza2));
        if (wave < 2)
          store_D(KC, 160, 128 + n0, mm_tile(AFS + 128, 192, Wdx2, 64, n0, 64, Bdx2));
      }
      __syncthreads();
      // RK4 combine
      const float cc = (e < 2) ? 0.5f * dt : ((e == 2) ? dt : 0.f);
      for (int i = tid; i < 16 * 160; i += NTHREADS) {
        float k = KC[i];
        if (e == 0) KA[i] = k;
        else        KA[i] += (e == 3 ? 1.f : 2.f) * k;
        if (e < 3)  UT[i] = U[i] + cc * k;
        else        U[i]  = U[i] + (dt / 6.f) * KA[i];
      }
      __syncthreads();
    }

    // ---- emit x at t+1 ----
    for (int i = tid; i < 16 * 32; i += NTHREADS) {
      int m = i >> 5, j = i & 31;
      out[((size_t)(bBase + m) * TT + (t + 1)) * XSD + j] = U[m * 160 + 128 + j];
    }
    __syncthreads();
  }
}

extern "C" void kernel_launch(void* const* d_in, const int* in_sizes, int n_in,
                              void* d_out, int out_size, void* d_ws, size_t ws_size,
                              hipStream_t stream) {
  (void)in_sizes; (void)n_in; (void)out_size; (void)d_ws; (void)ws_size;
  const float* x     = (const float*)d_in[0];
  const float* y     = (const float*)d_in[1];
  const float* ts    = (const float*)d_in[2];
  const float* a_wih = (const float*)d_in[3];
  const float* a_bih = (const float*)d_in[4];
  const float* a_whh = (const float*)d_in[5];
  const float* a_bhh = (const float*)d_in[6];
  const float* x_wih = (const float*)d_in[7];
  const float* x_bih = (const float*)d_in[8];
  const float* x_whh = (const float*)d_in[9];
  const float* x_bhh = (const float*)d_in[10];
  const float* zx_w1 = (const float*)d_in[11];
  const float* zx_b1 = (const float*)d_in[12];
  const float* zx_w2 = (const float*)d_in[13];
  const float* zx_b2 = (const float*)d_in[14];
  const float* za_w1 = (const float*)d_in[15];
  const float* za_b1 = (const float*)d_in[16];
  const float* za_w2 = (const float*)d_in[17];
  const float* za_b2 = (const float*)d_in[18];
  const float* dx_w1 = (const float*)d_in[19];
  const float* dx_b1 = (const float*)d_in[20];
  const float* dx_w2 = (const float*)d_in[21];
  const float* dx_b2 = (const float*)d_in[22];

  node_gru_ode_kernel<<<dim3(BB / ROWS), dim3(NTHREADS), SMEM_BYTES, stream>>>(
      x, y, ts,
      a_wih, a_bih, a_whh, a_bhh,
      x_wih, x_bih, x_whh, x_bhh,
      zx_w1, zx_b1, zx_w2, zx_b2,
      za_w1, za_b1, za_w2, za_b2,
      dx_w1, dx_b1, dx_w2, dx_b2,
      (float*)d_out);
}
